// SA_48533130445090
// MI455X (gfx1250) — matrix operand order repaired and checked by reading, not compiled, not run on hardware
//
#include <hip/hip_runtime.h>

// Self-attention (B=4, S=4096, D=64), fused flash-attention for gfx1250.
// Round 4: bit-packed masks (ballot pre-pass), f16 pre-converted x (xh row-major
// + xt transposed) in workspace, tile staging via global_load_async_to_lds_b128
// (ASYNCcnt DMA), base-2 softmax with log2e folded into Q.

#define Bn 4
#define Sn 4096
#define Dn 64
#define QT 64          // query rows per block (16 per wave x 4 waves)
#define KT 64          // keys per tile
#define NKT (Sn / KT)
#define NEGV   (-1.0e9f)
#define QSCALE (0.125f * 1.44269504088896340736f)  // 1/sqrt(64) * log2(e)
#define STR 72          // LDS row stride in halfs (144B: 16B-aligned, bank-spread)
#define SW  (Sn / 32)   // mask words per row

typedef __attribute__((ext_vector_type(16))) _Float16 v16h;
typedef __attribute__((ext_vector_type(8)))  _Float16 v8h;
typedef __attribute__((ext_vector_type(8)))  float    v8f;

union V16U { v16h v; v8h h[2]; _Float16 e[16]; };
union V8HU { v8h  h; _Float16 e[8]; };

static __device__ __forceinline__ v8f splat8(float x) {
    v8f r;
#pragma unroll
    for (int i = 0; i < 8; ++i) r[i] = x;
    return r;
}

static __device__ __forceinline__ float fexp2(float x) {
#if __has_builtin(__builtin_amdgcn_exp2f)
    return __builtin_amdgcn_exp2f(x);
#else
    return exp2f(x);
#endif
}

// Issue one 16-byte async DMA: global -> LDS (per active lane).
static __device__ __forceinline__ void async_b128(const _Float16* lds_dst,
                                                  const _Float16* gsrc) {
    const unsigned int  d = (unsigned int)(unsigned long long)(uintptr_t)lds_dst;
    const unsigned long long s = (unsigned long long)(uintptr_t)gsrc;
    asm volatile("global_load_async_to_lds_b128 %0, %1, off"
                 :: "v"(d), "v"(s) : "memory");
}

// ---------- pre-pass 1: pack attn_mask rows into bits (1 wave -> 32 keys) ----------
__global__ __launch_bounds__(256) void pack_amask_kernel(
    const int* __restrict__ am, unsigned int* __restrict__ abits) {
    const int row  = blockIdx.x;
    const int word = blockIdx.y * 8 + (threadIdx.x >> 5);
    const int lane = threadIdx.x & 31;
    const unsigned long long m = __ballot(am[(size_t)row * Sn + word * 32 + lane] != 0);
    if (lane == 0) abits[(size_t)row * SW + word] = (unsigned int)m;
}

// ---------- pre-pass 2: pack pad_mask into bits ----------
__global__ __launch_bounds__(256) void pack_pmask_kernel(
    const int* __restrict__ pm, unsigned int* __restrict__ pbits) {
    const int word = blockIdx.x * 8 + (threadIdx.x >> 5);   // 0 .. B*SW-1
    const int lane = threadIdx.x & 31;
    const unsigned long long m = __ballot(pm[(size_t)word * 32 + lane] != 0);
    if (lane == 0) pbits[word] = (unsigned int)m;
}

// ---------- pre-pass 3: x f32 -> f16 row-major (xh) + transposed (xt) ----------
__global__ __launch_bounds__(256) void cvt_x_kernel(
    const float* __restrict__ x, _Float16* __restrict__ xh, _Float16* __restrict__ xt) {
    const size_t i = ((size_t)blockIdx.x * 256 + threadIdx.x) * 4;  // over B*S*D
    const int b = (int)(i >> 18);          // S*D = 262144
    const int rem = (int)(i & 262143);
    const int s = rem >> 6;
    const int d = rem & 63;
    const float4 f = *(const float4*)(x + i);
    union { _Float16 e[4]; uint2 u; } hh;
    hh.e[0] = (_Float16)f.x; hh.e[1] = (_Float16)f.y;
    hh.e[2] = (_Float16)f.z; hh.e[3] = (_Float16)f.w;
    *(uint2*)(xh + i) = hh.u;
    _Float16* tb = xt + ((size_t)b * Dn + d) * Sn + s;
    tb[0 * Sn] = hh.e[0];
    tb[1 * Sn] = hh.e[1];
    tb[2 * Sn] = hh.e[2];
    tb[3 * Sn] = hh.e[3];
}

// ---------- main fused attention kernel ----------
__global__ __launch_bounds__(128) void SA_flash_wmma_kernel(
    const float*        __restrict__ x,      // [B, S, D] fp32 (for Q pre-scale)
    const _Float16*     __restrict__ xh,     // [B, S, D] f16
    const _Float16*     __restrict__ xt,     // [B, D, S] f16
    const unsigned int* __restrict__ abits,  // [S, S/32]
    const unsigned int* __restrict__ pbits,  // [B, S/32]
    float*              __restrict__ out)    // [B, S, D] fp32
{
    __shared__ __attribute__((aligned(16))) _Float16 lds_k [2][KT * STR];   // [buf][key][d]
    __shared__ __attribute__((aligned(16))) _Float16 lds_vt[2][Dn * STR];   // [buf][d][key]
    __shared__ __attribute__((aligned(16))) _Float16 lds_p [4 * 16 * STR];  // per-wave P^T [q][key]

    const int b      = blockIdx.y;
    const int qt     = blockIdx.x;
    const int tid    = threadIdx.x;
    const int w      = tid >> 5;
    const int lane   = tid & 31;
    const int lane16 = lane & 15;      // this lane's query (within wave's 16)
    const int hi     = lane >> 4;      // lane half

    const float*    xb  = x  + (size_t)b * Sn * Dn;
    const _Float16* xhb = xh + (size_t)b * Sn * Dn;
    const _Float16* xtb = xt + (size_t)b * Dn * Sn;
    const int q0 = qt * QT + w * 16;

    // ---- Q^T as gemm1 B operand, pre-scaled by log2e/sqrt(D) ----
    V16U bq[2];
    {
        const float* qrow = xb + (size_t)(q0 + lane16) * Dn;
#pragma unroll
        for (int c = 0; c < 2; ++c) {
#pragma unroll
            for (int j = 0; j < 2; ++j) {
                const float4 f0 = *(const float4*)(qrow + c * 32 + 8 * hi + j * 16);
                const float4 f1 = *(const float4*)(qrow + c * 32 + 8 * hi + j * 16 + 4);
                bq[c].e[j * 8 + 0] = (_Float16)(f0.x * QSCALE);
                bq[c].e[j * 8 + 1] = (_Float16)(f0.y * QSCALE);
                bq[c].e[j * 8 + 2] = (_Float16)(f0.z * QSCALE);
                bq[c].e[j * 8 + 3] = (_Float16)(f0.w * QSCALE);
                bq[c].e[j * 8 + 4] = (_Float16)(f1.x * QSCALE);
                bq[c].e[j * 8 + 5] = (_Float16)(f1.y * QSCALE);
                bq[c].e[j * 8 + 6] = (_Float16)(f1.z * QSCALE);
                bq[c].e[j * 8 + 7] = (_Float16)(f1.w * QSCALE);
            }
        }
    }

    const unsigned int* arb = abits + (size_t)(q0 + lane16) * SW;
    const unsigned int* prb = pbits + b * SW;

    // Async staging geometry: 2 chunks of 16 halfs per thread per array.
    const int ck0 = tid;             // chunks tid and tid+128 (of 256)
    // chunk ck: r = ck>>2 (row 0..63), seg = ck&3 (16-half segment)

    v8f o[4];
#pragma unroll
    for (int t = 0; t < 4; ++t) o[t] = splat8(0.0f);
    float mrow = -INFINITY;
    float lrow = 0.0f;

    _Float16* pldr = &lds_p[w * 16 * STR] + lane16 * STR;  // this lane's P^T row

    // Prologue: async-stage tile 0 into buffer 0.
#pragma unroll
    for (int j = 0; j < 2; ++j) {
        const int ck = ck0 + 128 * j, r = ck >> 2, seg = (ck & 3) * 16;
        async_b128(&lds_k [0][r * STR + seg], xhb + (size_t)r * Dn + seg);
        async_b128(&lds_vt[0][r * STR + seg], xtb + (size_t)r * Sn + seg);
    }

    for (int kt = 0; kt < NKT; ++kt) {
        const int kbase = kt * KT;
        const int cur = kt & 1;
        const int nxt = cur ^ 1;

        asm volatile("s_wait_asynccnt 0" ::: "memory");  // own DMAs done
        __syncthreads();                                 // everyone's DMAs done

        // ---- async-stage tile kt+1 (DMA overlaps all compute below) ----
        if (kt + 1 < NKT) {
            const int nbase = kbase + KT;
#pragma unroll
            for (int j = 0; j < 2; ++j) {
                const int ck = ck0 + 128 * j, r = ck >> 2, seg = (ck & 3) * 16;
                async_b128(&lds_k [nxt][r * STR + seg], xhb + (size_t)(nbase + r) * Dn + seg);
                async_b128(&lds_vt[nxt][r * STR + seg], xtb + (size_t)r * Sn + nbase + seg);
            }
        }

        // ---- S^T = K (Q*log2e/sqrtD)^T : A = key tile (LDS), B = bq (regs) ----
        v8f s[4];
#pragma unroll
        for (int t = 0; t < 4; ++t) s[t] = splat8(0.0f);
#pragma unroll
        for (int c = 0; c < 2; ++c) {
#pragma unroll
            for (int t = 0; t < 4; ++t) {
                const _Float16* ap = &lds_k[cur][(t * 16 + lane16) * STR + c * 32 + 8 * hi];
                V16U av;
                av.h[0] = *(const v8h*)ap;
                av.h[1] = *(const v8h*)(ap + 16);
                s[t] = __builtin_amdgcn_wmma_f32_16x16x32_f16(
                    false, av.v, false, bq[c].v, (short)0, s[t], false, false);
            }
        }

        // ---- masks from packed bits: 8 bytes per lane per tile ----
        {
            const uint2 aw = *(const uint2*)(arb + (kbase >> 5));
            const uint2 pw = *(const uint2*)(prb + (kbase >> 5));
            const unsigned int cw0 = aw.x & pw.x;
            const unsigned int cw1 = aw.y & pw.y;
#pragma unroll
            for (int t = 0; t < 4; ++t) {
                const unsigned int byt =
                    ((t & 2) ? cw1 : cw0) >> (((t & 1) << 4) + 8 * hi);
#pragma unroll
                for (int v = 0; v < 8; ++v)
                    s[t][v] = (byt & (1u << v)) ? s[t][v] : NEGV;
            }
        }

        // ---- online softmax (base-2), scalar per lane ----
        float tmax = s[0][0];
#pragma unroll
        for (int t = 0; t < 4; ++t)
#pragma unroll
            for (int v = 0; v < 8; ++v) tmax = fmaxf(tmax, s[t][v]);
        tmax = fmaxf(tmax, __shfl_xor(tmax, 16, 32));

        const float mnew = fmaxf(mrow, tmax);
        const float corr = fexp2(mrow - mnew);
        mrow = mnew;

        float rsum = 0.0f;
#pragma unroll
        for (int t = 0; t < 4; ++t)
#pragma unroll
            for (int v = 0; v < 8; ++v) {
                const float pv = fexp2(s[t][v] - mnew);
                s[t][v] = pv;
                rsum += pv;
            }
        rsum += __shfl_xor(rsum, 16, 32);
        lrow = lrow * corr + rsum;
#pragma unroll
        for (int t = 0; t < 4; ++t)
#pragma unroll
            for (int v = 0; v < 8; ++v) o[t][v] *= corr;

        // ---- P^T staging: one b128 store per tile, reload as B operand ----
#pragma unroll
        for (int t = 0; t < 4; ++t) {
            V8HU pk;
#pragma unroll
            for (int v = 0; v < 8; ++v) pk.e[v] = (_Float16)s[t][v];
            *(v8h*)(pldr + t * 16 + 8 * hi) = pk.h;
        }
        V16U pb[2];
#pragma unroll
        for (int c = 0; c < 2; ++c) {
            pb[c].h[0] = *(const v8h*)(pldr + c * 32 + 8 * hi);
            pb[c].h[1] = *(const v8h*)(pldr + c * 32 + 8 * hi + 16);
        }

        // ---- O^T += V^T P^T ----
#pragma unroll
        for (int c = 0; c < 2; ++c) {
#pragma unroll
            for (int dt = 0; dt < 4; ++dt) {
                const _Float16* ap = &lds_vt[cur][(dt * 16 + lane16) * STR + c * 32 + 8 * hi];
                V16U av;
                av.h[0] = *(const v8h*)ap;
                av.h[1] = *(const v8h*)(ap + 16);
                o[dt] = __builtin_amdgcn_wmma_f32_16x16x32_f16(
                    false, av.v, false, pb[c].v, (short)0, o[dt], false, false);
            }
        }
    }

    // ---- epilogue ----
    {
        const float inv = __builtin_amdgcn_rcpf(lrow);
        float* orow = out + ((size_t)b * Sn + q0 + lane16) * Dn;
#pragma unroll
        for (int dt = 0; dt < 4; ++dt) {
            float4 f0, f1;
            f0.x = o[dt][0] * inv; f0.y = o[dt][1] * inv;
            f0.z = o[dt][2] * inv; f0.w = o[dt][3] * inv;
            f1.x = o[dt][4] * inv; f1.y = o[dt][5] * inv;
            f1.z = o[dt][6] * inv; f1.w = o[dt][7] * inv;
            *(float4*)(orow + dt * 16 + 8 * hi)     = f0;
            *(float4*)(orow + dt * 16 + 8 * hi + 4) = f1;
        }
    }
}

extern "C" void kernel_launch(void* const* d_in, const int* in_sizes, int n_in,
                              void* d_out, int out_size, void* d_ws, size_t ws_size,
                              hipStream_t stream) {
    (void)in_sizes; (void)n_in; (void)out_size; (void)ws_size;
    const float* x  = (const float*)d_in[0];
    const int*   am = (const int*)d_in[1];
    const int*   pm = (const int*)d_in[2];
    float*       o  = (float*)d_out;

    // Workspace layout (bytes): xh 2MB | xt 2MB | abits 2MB | pbits 2KB
    char* ws = (char*)d_ws;
    _Float16*     xh    = (_Float16*)(ws);
    _Float16*     xt    = (_Float16*)(ws + (2u << 20));
    unsigned int* abits = (unsigned int*)(ws + (4u << 20));
    unsigned int* pbits = (unsigned int*)(ws + (6u << 20));

    cvt_x_kernel<<<(Bn * Sn * Dn) / 1024, 256, 0, stream>>>(x, xh, xt);
    pack_amask_kernel<<<dim3(Sn, SW / 8), 256, 0, stream>>>(am, abits);
    pack_pmask_kernel<<<(Bn * SW) / 8, 256, 0, stream>>>(pm, pbits);

    dim3 grid(Sn / QT, Bn);
    SA_flash_wmma_kernel<<<grid, 128, 0, stream>>>(x, xh, xt, abits, pbits, o);
}
